// E75VectorGateCell_79946521247811
// MI455X (gfx1250) — compile-verified, matrix operand order checked by reading, not run on hardware
//
#include <hip/hip_runtime.h>
#include <math.h>

// Problem constants (from reference)
#define T_STEPS 2048
#define BATCH   16
#define DIM     1024
#define NDIM    64
#define M_ROWS  (T_STEPS * BATCH)   // 32768 = GEMM M
#define NCOLS   256                 // 4 matrices x 64 = GEMM N
#define EPSV    1e-6f

typedef __attribute__((ext_vector_type(16))) __bf16 v16bf;
typedef __attribute__((ext_vector_type(8)))  __bf16 v8bf;
typedef __attribute__((ext_vector_type(8)))  float  v8f;
typedef __attribute__((ext_vector_type(4)))  float  v4f;

// ---------------------------------------------------------------------------
// fp32 -> bf16 streaming conversion (8 elements / thread)
// ---------------------------------------------------------------------------
__global__ __launch_bounds__(256) void cvt_f32_bf16_kernel(
    const float* __restrict__ src, __bf16* __restrict__ dst, int n8)
{
    int tid = blockIdx.x * blockDim.x + threadIdx.x;
    if (tid >= n8) return;
    const v4f* s = (const v4f*)(src + (size_t)tid * 8);
    v4f a = s[0], b = s[1];
    v8bf o;
    o[0] = (__bf16)a[0]; o[1] = (__bf16)a[1]; o[2] = (__bf16)a[2]; o[3] = (__bf16)a[3];
    o[4] = (__bf16)b[0]; o[5] = (__bf16)b[1]; o[6] = (__bf16)b[2]; o[7] = (__bf16)b[3];
    *(v8bf*)(dst + (size_t)tid * 8) = o;
}

// ---------------------------------------------------------------------------
// Projection GEMM: [M=32768, K=1024] x [N=256, K=1024]^T -> k|v|q|g fp32
// One 16-row M tile per block; wave w computes N-tiles 2w and 2w+1 with
// v_wmma_f32_16x16x32_bf16, 32 K-steps. Sigmoid fused for the beta matrix.
//
// 16-bit A/B fragment layout (ISA 7.12.2): lane<16 holds K {k..k+7, k+16..k+23},
// lane>=16 holds K {k+8..k+15, k+24..k+31}; B column-major mirrors A.
// ---------------------------------------------------------------------------
__global__ __launch_bounds__(256) void proj_wmma_kernel(
    const __bf16* __restrict__ xbf,    // [M, DIM]
    const __bf16* __restrict__ wbf,    // [256, DIM]  rows: Wk|Wv|Wq|Wb
    const float*  __restrict__ b_beta, // [64]
    float* __restrict__ k_arr, float* __restrict__ v_arr,
    float* __restrict__ q_arr, float* __restrict__ g_arr)
{
    const int lane    = threadIdx.x & 31;
    const int wave    = threadIdx.x >> 5;          // 0..7
    const int lid     = lane & 15;
    const int halfsel = lane >> 4;                 // 0/1
    const int koff    = halfsel * 8;               // K sub-offset in 32-block
    const int m0      = blockIdx.x * 16;

    const int nt0 = wave * 2, nt1 = nt0 + 1;       // N-tiles (same matrix)
    const int mat = nt0 >> 2;                      // 0..3 -> k,v,q,beta
    const int nc0 = (nt0 & 3) * 16 + lid;          // column within matrix
    const int nc1 = (nt1 & 3) * 16 + lid;

    const __bf16* arow  = xbf + (size_t)(m0 + lid) * DIM;
    const __bf16* brow0 = wbf + (size_t)(mat * 64 + nc0) * DIM;
    const __bf16* brow1 = wbf + (size_t)(mat * 64 + nc1) * DIM;

    v8f acc0 = {}, acc1 = {};

    #pragma unroll 4
    for (int k = 0; k < DIM; k += 32) {
        v8bf alo = *(const v8bf*)(arow  + k + koff);
        v8bf ahi = *(const v8bf*)(arow  + k + 16 + koff);
        v8bf b0l = *(const v8bf*)(brow0 + k + koff);
        v8bf b0h = *(const v8bf*)(brow0 + k + 16 + koff);
        v8bf b1l = *(const v8bf*)(brow1 + k + koff);
        v8bf b1h = *(const v8bf*)(brow1 + k + 16 + koff);
        v16bf A, B0, B1;
        #pragma unroll
        for (int i = 0; i < 8; ++i) {
            A[i] = alo[i];  A[8 + i] = ahi[i];
            B0[i] = b0l[i]; B0[8 + i] = b0h[i];
            B1[i] = b1l[i]; B1[8 + i] = b1h[i];
        }
        acc0 = __builtin_amdgcn_wmma_f32_16x16x32_bf16(false, A, false, B0,
                                                       (short)0, acc0, false, false);
        acc1 = __builtin_amdgcn_wmma_f32_16x16x32_bf16(false, A, false, B1,
                                                       (short)0, acc1, false, false);
    }

    float* dst = (mat == 0) ? k_arr : (mat == 1) ? v_arr : (mat == 2) ? q_arr : g_arr;
    float bb0 = 0.f, bb1 = 0.f;
    if (mat == 3) { bb0 = b_beta[nc0]; bb1 = b_beta[nc1]; }

    // C layout: VGPR i holds M = 8*halfsel + i, N = lid (within tile)
    #pragma unroll
    for (int i = 0; i < 8; ++i) {
        int m = halfsel * 8 + i;
        float x0 = acc0[i], x1 = acc1[i];
        if (mat == 3) {
            x0 = 1.f / (1.f + __expf(-(x0 + bb0)));
            x1 = 1.f / (1.f + __expf(-(x1 + bb1)));
        }
        dst[(size_t)(m0 + m) * NDIM + nc0] = x0;
        dst[(size_t)(m0 + m) * NDIM + nc1] = x1;
    }
}

// ---------------------------------------------------------------------------
// Sequential gated delta-rule scan. Lane = 4*row_local + col_chunk:
// each lane owns S[r][16c..16c+15] in registers. All reductions are within a
// 4-lane group (shfl_xor 1,2): no LDS, no barriers in the 2048-step loop.
// Next step's k/q/v/g are register-prefetched; +2 steps via global_prefetch.
// ---------------------------------------------------------------------------
__global__ __launch_bounds__(256) void scan_kernel(
    const float* __restrict__ k_arr, const float* __restrict__ v_arr,
    const float* __restrict__ q_arr, const float* __restrict__ g_arr,
    const float* __restrict__ S0,
    float* __restrict__ out,       // [T,B,N]
    float* __restrict__ s_final)   // [B,N,N]
{
    const int b    = blockIdx.x;          // batch
    const int wave = threadIdx.x >> 5;    // row-group 0..7 (independent)
    const int lane = threadIdx.x & 31;
    const int rl   = lane >> 2;           // row within group
    const int c    = lane & 3;            // 16-col chunk
    const int r    = wave * 8 + rl;       // 0..63
    const int j0   = c * 16;

    // State: 16 columns of row r
    float S[16];
    #pragma unroll
    for (int u = 0; u < 4; ++u) {
        v4f s0 = *(const v4f*)(S0 + ((size_t)b * NDIM + r) * NDIM + j0 + 4 * u);
        S[4*u+0] = s0[0]; S[4*u+1] = s0[1]; S[4*u+2] = s0[2]; S[4*u+3] = s0[3];
    }

    const size_t strideT = (size_t)BATCH * NDIM;
    const float* kb = k_arr + (size_t)b * NDIM;
    const float* qb = q_arr + (size_t)b * NDIM;
    const float* vb = v_arr + (size_t)b * NDIM;
    const float* gb = g_arr + (size_t)b * NDIM;

    // Prefetched operands for step t (loaded one iteration ahead)
    v4f pk[4], pq[4]; float pv, pg;
    {
        const v4f* kv = (const v4f*)(kb + j0);
        const v4f* qv = (const v4f*)(qb + j0);
        #pragma unroll
        for (int u = 0; u < 4; ++u) { pk[u] = kv[u]; pq[u] = qv[u]; }
        pv = vb[r]; pg = gb[r];
    }

    for (int t = 0; t < T_STEPS; ++t) {
        // consume prefetched operands
        v4f ck[4], cq[4];
        #pragma unroll
        for (int u = 0; u < 4; ++u) { ck[u] = pk[u]; cq[u] = pq[u]; }
        float cv = pv, cg = pg;

        // issue next step's loads before computing (latency hidden by compute)
        if (t + 1 < T_STEPS) {
            const size_t off = (size_t)(t + 1) * strideT;
            const v4f* kv = (const v4f*)(kb + off + j0);
            const v4f* qv = (const v4f*)(qb + off + j0);
            #pragma unroll
            for (int u = 0; u < 4; ++u) { pk[u] = kv[u]; pq[u] = qv[u]; }
            pv = vb[off + r]; pg = gb[off + r];
        }
        if (t + 2 < T_STEPS) {
            const size_t off2 = (size_t)(t + 2) * strideT;
            __builtin_prefetch(kb + off2 + j0, 0, 1);
            __builtin_prefetch(qb + off2 + j0, 0, 1);
        }

        // ||k||^2 (partial over this lane's 16 cols) and S·k partial dot
        float ss = 0.f, rdot = 0.f;
        #pragma unroll
        for (int u = 0; u < 4; ++u)
            #pragma unroll
            for (int e = 0; e < 4; ++e) {
                float kk = ck[u][e];
                ss   = fmaf(kk, kk, ss);
                rdot = fmaf(S[4*u+e], kk, rdot);
            }
        ss   += __shfl_xor(ss,   1, 32);  ss   += __shfl_xor(ss,   2, 32);
        rdot += __shfl_xor(rdot, 1, 32);  rdot += __shfl_xor(rdot, 2, 32);

        float inv       = 1.f / (sqrtf(ss) + EPSV);   // 1/(||k||+eps)
        float retrieved = rdot * inv;                 // (S·k)·inv = S·k_norm
        float scale     = (cv - retrieved) * inv;     // delta * inv

        // S = g*S + scale*k ; accumulate S_new · q
        float qdot = 0.f;
        #pragma unroll
        for (int u = 0; u < 4; ++u)
            #pragma unroll
            for (int e = 0; e < 4; ++e) {
                float sn = fmaf(cg, S[4*u+e], scale * ck[u][e]);
                S[4*u+e] = sn;
                qdot = fmaf(sn, cq[u][e], qdot);
            }
        qdot += __shfl_xor(qdot, 1, 32);  qdot += __shfl_xor(qdot, 2, 32);

        if (c == 0) {
            float sq  = qdot;
            float sig = 1.f / (1.f + __expf(-sq));
            out[(size_t)t * strideT + (size_t)b * NDIM + r] = sq * sq * sig;
        }
    }

    // S_final
    #pragma unroll
    for (int u = 0; u < 4; ++u) {
        v4f sv; sv[0] = S[4*u+0]; sv[1] = S[4*u+1]; sv[2] = S[4*u+2]; sv[3] = S[4*u+3];
        *(v4f*)(s_final + ((size_t)b * NDIM + r) * NDIM + j0 + 4 * u) = sv;
    }
}

// ---------------------------------------------------------------------------
// Host-side launch
// ---------------------------------------------------------------------------
extern "C" void kernel_launch(void* const* d_in, const int* in_sizes, int n_in,
                              void* d_out, int out_size, void* d_ws, size_t ws_size,
                              hipStream_t stream)
{
    const float* x      = (const float*)d_in[0];  // [T,B,DIM]
    const float* S0     = (const float*)d_in[1];  // [B,N,N]
    const float* W_k    = (const float*)d_in[2];  // [N,DIM]
    const float* W_v    = (const float*)d_in[3];
    const float* W_q    = (const float*)d_in[4];
    const float* W_beta = (const float*)d_in[5];
    const float* b_beta = (const float*)d_in[6];  // [N]

    float* out_seq = (float*)d_out;                         // [T,B,N]
    float* s_final = out_seq + (size_t)T_STEPS * BATCH * NDIM;

    // Workspace layout
    char* ws = (char*)d_ws;
    __bf16* xbf  = (__bf16*)ws;                             // M*DIM bf16
    size_t  off  = (size_t)M_ROWS * DIM * sizeof(__bf16);
    __bf16* wbf  = (__bf16*)(ws + off);                     // 256*DIM bf16
    off += (size_t)NCOLS * DIM * sizeof(__bf16);
    float* k_arr = (float*)(ws + off); off += (size_t)M_ROWS * NDIM * sizeof(float);
    float* v_arr = (float*)(ws + off); off += (size_t)M_ROWS * NDIM * sizeof(float);
    float* q_arr = (float*)(ws + off); off += (size_t)M_ROWS * NDIM * sizeof(float);
    float* g_arr = (float*)(ws + off);

    // 1) bf16 conversions
    {
        int n8 = (M_ROWS * DIM) / 8;
        cvt_f32_bf16_kernel<<<(n8 + 255) / 256, 256, 0, stream>>>(x, xbf, n8);
        int w8 = (NDIM * DIM) / 8;
        int wb = (w8 + 255) / 256;
        cvt_f32_bf16_kernel<<<wb, 256, 0, stream>>>(W_k,    wbf + 0 * (size_t)NDIM * DIM, w8);
        cvt_f32_bf16_kernel<<<wb, 256, 0, stream>>>(W_v,    wbf + 1 * (size_t)NDIM * DIM, w8);
        cvt_f32_bf16_kernel<<<wb, 256, 0, stream>>>(W_q,    wbf + 2 * (size_t)NDIM * DIM, w8);
        cvt_f32_bf16_kernel<<<wb, 256, 0, stream>>>(W_beta, wbf + 3 * (size_t)NDIM * DIM, w8);
    }

    // 2) WMMA projection GEMM (+ fused sigmoid for beta)
    proj_wmma_kernel<<<M_ROWS / 16, 256, 0, stream>>>(
        xbf, wbf, b_beta, k_arr, v_arr, q_arr, g_arr);

    // 3) Sequential scan: 16 blocks (one per batch), 8 independent waves each
    scan_kernel<<<BATCH, 256, 0, stream>>>(
        k_arr, v_arr, q_arr, g_arr, S0, out_seq, s_final);
}
